// CrossAttention_20435454394452
// MI455X (gfx1250) — compile-verified, matrix operand order checked by reading, not run on hardware
//
#include <hip/hip_runtime.h>

// ---------------------------------------------------------------------------
// Types for CDNA5 WMMA (wave32, 16x16x32 bf16 -> f32)
// ---------------------------------------------------------------------------
typedef __attribute__((ext_vector_type(16))) __bf16 v16bf;
typedef __attribute__((ext_vector_type(8)))  float  v8f;

// fp32 -> bf16 round-to-nearest-even
__device__ __forceinline__ unsigned short f2bf(float x) {
    union { float f; unsigned u; } v; v.f = x;
    unsigned r = v.u + 0x7FFFu + ((v.u >> 16) & 1u);
    return (unsigned short)(r >> 16);
}

// Load a 16x32 bf16 fragment from an LDS tile laid out row-major
// (row stride = stride bytes, K contiguous), per the CDNA5 ISA layout:
//   lanes 0-15 : row = lane,      elems 0-7 -> K=k0+0..7,  elems 8-15 -> K=k0+16..23
//   lanes 16-31: row = lane-16,   elems 0-7 -> K=k0+8..15, elems 8-15 -> K=k0+24..31
// Used for both A fragments (row = M) and B fragments (row = N, tile holds B^T).
__device__ __forceinline__ v16bf ld_frag(const char* tile, int stride, int k0, int lane) {
    int r  = lane & 15;
    int kb = (lane >> 4) << 3;
    const char* p = tile + r * stride + (k0 + kb) * 2;
    union { uint4 q[2]; v16bf v; } u;
    u.q[0] = *(const uint4*)(p);        // K = k0+kb+0..7   (2 bf16 per dword)
    u.q[1] = *(const uint4*)(p + 32);   // K = k0+kb+16..23
    return u.v;
}

// ===========================================================================
// Projection GEMM:  Y[m][n] = sum_k X[m][k] * W[n][k] + bias[n]   (Y in bf16)
// M x 768 x 768, workgroup tile 128(M) x 64(N), 8 waves, each wave 16x64.
// K-step 64 -> 8 WMMAs per wave between barriers.
// ===========================================================================
#define PJ_BM 128
#define PJ_BN 64
#define PJ_AS 144         // (64+8)*2 bytes padded LDS row stride
#define EMBD 768

__global__ __launch_bounds__(256)
void proj_kernel(const float* __restrict__ X, const float* __restrict__ W,
                 const float* __restrict__ bias, unsigned short* __restrict__ Y) {
    __shared__ char lds[(PJ_BM + PJ_BN) * PJ_AS];
    char* lds_a = lds;
    char* lds_w = lds + PJ_BM * PJ_AS;

    const int tid   = threadIdx.x;
    const int lane  = tid & 31;
    const int wv    = tid >> 5;
    const int mbase = blockIdx.x * PJ_BM;
    const int nbase = blockIdx.y * PJ_BN;

    v8f acc[4] = {};

    for (int k0 = 0; k0 < EMBD; k0 += 64) {
        // stage A tile: 128 rows x 32 dword-pairs (fp32 -> bf16)
        for (int i = tid; i < PJ_BM * 32; i += 256) {
            int row = i >> 5, kp = i & 31;
            const float* src = X + (size_t)(mbase + row) * EMBD + k0 + kp * 2;
            float2 xv = *(const float2*)src;
            if (k0 + 64 < EMBD) __builtin_prefetch(src + 64, 0, 0);  // global_prefetch_b8
            unsigned pk = (unsigned)f2bf(xv.x) | ((unsigned)f2bf(xv.y) << 16);
            *(unsigned*)(lds_a + row * PJ_AS + kp * 4) = pk;
        }
        // stage W tile: 64 rows x 32 pairs
        for (int i = tid; i < PJ_BN * 32; i += 256) {
            int row = i >> 5, kp = i & 31;
            float2 wvv = *(const float2*)(W + (size_t)(nbase + row) * EMBD + k0 + kp * 2);
            unsigned pk = (unsigned)f2bf(wvv.x) | ((unsigned)f2bf(wvv.y) << 16);
            *(unsigned*)(lds_w + row * PJ_AS + kp * 4) = pk;
        }
        __syncthreads();

#pragma unroll
        for (int ks = 0; ks < 2; ++ks) {
            v16bf a = ld_frag(lds_a + wv * 16 * PJ_AS, PJ_AS, ks * 32, lane);
#pragma unroll
            for (int f = 0; f < 4; ++f) {
                v16bf bb = ld_frag(lds_w + f * 16 * PJ_AS, PJ_AS, ks * 32, lane);
                acc[f] = __builtin_amdgcn_wmma_f32_16x16x32_bf16(
                    false, a, false, bb, (short)0, acc[f], false, false);
            }
        }
        __syncthreads();
    }

    // epilogue: + bias, convert to bf16, store
    const int col0  = lane & 15;
    const int rhalf = (lane >> 4) * 8;
#pragma unroll
    for (int f = 0; f < 4; ++f) {
        int col  = nbase + f * 16 + col0;
        float bv = bias[col];
#pragma unroll
        for (int v = 0; v < 8; ++v) {
            size_t m = (size_t)(mbase + wv * 16 + rhalf + v);
            Y[m * EMBD + col] = f2bf(acc[f][v] + bv);
        }
    }
}

// ===========================================================================
// Flash-style cross attention.
// One workgroup (8 waves) per (batch, 16 Q-rows). Online softmax over L in
// chunks of 64. E-reduction for S and E-columns of O are split across waves
// (96 cols each); partial S merged with ds_add_f32. Softmax is parallel:
// wave w owns rows 2w,2w+1; 16 lanes per row, shfl_xor reductions.
// ===========================================================================
#define AT_LC  64
#define AT_QS  1552                 // (768+8)*2
#define AT_KS  1552
#define AT_VS  144                  // (64+8)*2
#define AT_PS  144
#define OFF_Q  0
#define OFF_KV (16 * AT_QS)                    // 24832
#define SZ_KV  (EMBD * AT_VS)                  // 110592 >= 64*1552
#define OFF_S  (OFF_KV + SZ_KV)                // S: 16x64 f32
#define OFF_P  (OFF_S + 16 * AT_LC * 4)        // P: 16 rows x 144B
#define OFF_AL (OFF_P + 16 * AT_PS)            // alpha[16] f32
#define LDS_ATTN (OFF_AL + 64)                 // 141888 bytes

__global__ __launch_bounds__(256)
void attn_kernel(const unsigned short* __restrict__ Qb,
                 const unsigned short* __restrict__ Kb,
                 const unsigned short* __restrict__ Vb,
                 float* __restrict__ out) {
    extern __shared__ char smem[];
    char*  lds_q  = smem + OFF_Q;
    char*  lds_kv = smem + OFF_KV;
    float* lds_s  = (float*)(smem + OFF_S);
    char*  lds_p  = smem + OFF_P;
    float* lds_al = (float*)(smem + OFF_AL);

    const int tid  = threadIdx.x;
    const int lane = tid & 31;
    const int wv   = tid >> 5;
    const int b    = blockIdx.x >> 7;          // 128 row-tiles per batch
    const int trow = blockIdx.x & 127;
    const size_t qrow0 = (size_t)b * 2048 + (size_t)trow * 16;
    const size_t krow0 = (size_t)b * 2048;
    const int eslice = wv * 96;
    const float scale = 0.0360843918243516f;   // 768^-0.5

    // stage Q tile (bf16 copy): 16 rows x 384 dwords
    for (int i = tid; i < 16 * 384; i += 256) {
        int r = i / 384, c = i % 384;
        *(unsigned*)(lds_q + r * AT_QS + c * 4) =
            *(const unsigned*)(Qb + (qrow0 + r) * EMBD + c * 2);
    }
    __syncthreads();

    // hoist this wave's 3 Q A-fragments (loop-invariant)
    v16bf qf[3];
#pragma unroll
    for (int ks = 0; ks < 3; ++ks) qf[ks] = ld_frag(lds_q, AT_QS, eslice + ks * 32, lane);

    // per-lane online-softmax state for this lane's row (row = wv*2 + (lane>>4))
    float m_r = -3.0e38f, l_r = 0.0f;
    v8f oacc[6] = {};

    for (int lc = 0; lc < 2048; lc += AT_LC) {
        // zero S, stage K chunk (64 x 768 bf16)
        for (int i = tid; i < 16 * AT_LC; i += 256) lds_s[i] = 0.0f;
        for (int i = tid; i < 64 * 384; i += 256) {
            int r = i / 384, c = i % 384;
            *(unsigned*)(lds_kv + r * AT_KS + c * 4) =
                *(const unsigned*)(Kb + (krow0 + lc + r) * EMBD + c * 2);
        }
        __syncthreads();

        // partial S over this wave's 96-wide E slice: 3 k-steps x 4 n-tiles
        v8f sacc[4] = {};
#pragma unroll
        for (int ks = 0; ks < 3; ++ks) {
#pragma unroll
            for (int f = 0; f < 4; ++f) {
                v16bf bb = ld_frag(lds_kv + f * 16 * AT_KS, AT_KS, eslice + ks * 32, lane);
                sacc[f] = __builtin_amdgcn_wmma_f32_16x16x32_bf16(
                    false, qf[ks], false, bb, (short)0, sacc[f], false, false);
            }
        }
        // merge partial S (scaled) into LDS via ds_add_f32
        {
            int col0 = lane & 15, rhalf = (lane >> 4) * 8;
#pragma unroll
            for (int f = 0; f < 4; ++f)
#pragma unroll
                for (int v = 0; v < 8; ++v)
                    atomicAdd(&lds_s[(rhalf + v) * AT_LC + f * 16 + col0],
                              sacc[f][v] * scale);
        }
        __syncthreads();

        // parallel online softmax: wave wv owns rows 2wv and 2wv+1
        {
            int r  = wv * 2 + (lane >> 4);
            int c0 = lane & 15;
            float s0 = lds_s[r * AT_LC + c0];
            float s1 = lds_s[r * AT_LC + c0 + 16];
            float s2 = lds_s[r * AT_LC + c0 + 32];
            float s3 = lds_s[r * AT_LC + c0 + 48];
            float mc = fmaxf(fmaxf(s0, s1), fmaxf(s2, s3));
#pragma unroll
            for (int off = 1; off < 16; off <<= 1)
                mc = fmaxf(mc, __shfl_xor(mc, off, 16));
            float mn = fmaxf(m_r, mc);
            float al = __expf(m_r - mn);
            float p0 = __expf(s0 - mn), p1 = __expf(s1 - mn);
            float p2 = __expf(s2 - mn), p3 = __expf(s3 - mn);
            float sum = (p0 + p1) + (p2 + p3);
#pragma unroll
            for (int off = 1; off < 16; off <<= 1)
                sum += __shfl_xor(sum, off, 16);
            l_r = l_r * al + sum;
            m_r = mn;
            if (c0 == 0) lds_al[r] = al;
            *(unsigned short*)(lds_p + r * AT_PS + c0 * 2)        = f2bf(p0);
            *(unsigned short*)(lds_p + r * AT_PS + (c0 + 16) * 2) = f2bf(p1);
            *(unsigned short*)(lds_p + r * AT_PS + (c0 + 32) * 2) = f2bf(p2);
            *(unsigned short*)(lds_p + r * AT_PS + (c0 + 48) * 2) = f2bf(p3);
        }
        __syncthreads();

        // rescale O accumulators by alpha (rows rhalf..rhalf+7 per lane half)
        {
            int rhalf = (lane >> 4) * 8;
            float a8[8];
#pragma unroll
            for (int v = 0; v < 8; ++v) a8[v] = lds_al[rhalf + v];
#pragma unroll
            for (int f = 0; f < 6; ++f)
#pragma unroll
                for (int v = 0; v < 8; ++v) oacc[f][v] *= a8[v];
        }
        // stage V transposed: lds_vt[e][l]  (reuses K buffer)
        for (int i = tid; i < 64 * 384; i += 256) {
            int e2 = i % 384, r = i / 384;
            unsigned v2 = *(const unsigned*)(Vb + (krow0 + lc + r) * EMBD + e2 * 2);
            *(unsigned short*)(lds_kv + (2 * e2)     * AT_VS + r * 2) = (unsigned short)(v2 & 0xffffu);
            *(unsigned short*)(lds_kv + (2 * e2 + 1) * AT_VS + r * 2) = (unsigned short)(v2 >> 16);
        }
        __syncthreads();

        // O += P x V  (this wave's 96 output columns): 2 k-steps x 6 n-tiles
#pragma unroll
        for (int ks = 0; ks < 2; ++ks) {
            v16bf a = ld_frag(lds_p, AT_PS, ks * 32, lane);
#pragma unroll
            for (int f = 0; f < 6; ++f) {
                v16bf bb = ld_frag(lds_kv + (eslice + f * 16) * AT_VS, AT_VS, ks * 32, lane);
                oacc[f] = __builtin_amdgcn_wmma_f32_16x16x32_bf16(
                    false, a, false, bb, (short)0, oacc[f], false, false);
            }
        }
        __syncthreads();   // protect lds_kv/lds_s before next iteration's staging
    }

    // final 1/l normalization and fp32 store
    {
        int r = wv * 2 + (lane >> 4);
        if ((lane & 15) == 0) lds_al[r] = 1.0f / l_r;
    }
    __syncthreads();
    {
        int col0 = lane & 15, rhalf = (lane >> 4) * 8;
        float li[8];
#pragma unroll
        for (int v = 0; v < 8; ++v) li[v] = lds_al[rhalf + v];
#pragma unroll
        for (int f = 0; f < 6; ++f) {
            int col = eslice + f * 16 + col0;
#pragma unroll
            for (int v = 0; v < 8; ++v) {
                size_t row = qrow0 + rhalf + v;
                out[row * EMBD + col] = oacc[f][v] * li[v];
            }
        }
    }
}

// ===========================================================================
extern "C" void kernel_launch(void* const* d_in, const int* in_sizes, int n_in,
                              void* d_out, int out_size, void* d_ws, size_t ws_size,
                              hipStream_t stream) {
    (void)in_sizes; (void)n_in; (void)out_size; (void)ws_size;
    const float* x   = (const float*)d_in[0];
    const float* ctx = (const float*)d_in[1];
    const float* Wq  = (const float*)d_in[2];
    const float* bq  = (const float*)d_in[3];
    const float* Wk  = (const float*)d_in[4];
    const float* bk  = (const float*)d_in[5];
    const float* Wv  = (const float*)d_in[6];
    const float* bv  = (const float*)d_in[7];
    float* out = (float*)d_out;

    const int M = 16 * 2048;                       // B*T == B*L
    unsigned short* Qb = (unsigned short*)d_ws;    // bf16 Q/K/V in workspace
    unsigned short* Kb = Qb + (size_t)M * EMBD;
    unsigned short* Vb = Kb + (size_t)M * EMBD;

    dim3 pgrid(M / PJ_BM, EMBD / PJ_BN);
    proj_kernel<<<pgrid, 256, 0, stream>>>(x,   Wq, bq, Qb);
    proj_kernel<<<pgrid, 256, 0, stream>>>(ctx, Wk, bk, Kb);
    proj_kernel<<<pgrid, 256, 0, stream>>>(ctx, Wv, bv, Vb);

    hipFuncSetAttribute((const void*)attn_kernel,
                        hipFuncAttributeMaxDynamicSharedMemorySize, LDS_ATTN);
    attn_kernel<<<16 * 2048 / 16, 256, LDS_ATTN, stream>>>(Qb, Kb, Vb, out);
}
